// TransformerDecoderLayer_50929722196891
// MI455X (gfx1250) — matrix operand order repaired and checked by reading, not compiled, not run on hardware
//
#include <hip/hip_runtime.h>

typedef __attribute__((ext_vector_type(16))) __bf16        v16bf;
typedef __attribute__((ext_vector_type(8)))  float         v8f;
typedef __attribute__((ext_vector_type(8)))  unsigned int  v8u;

#define D_MODEL 1024
#define N_HEAD  16
#define D_FF    4096
#define BATCH   4
#define SEQ     1024
#define DK      64
#define NEGBIG  (-1e20f)

__device__ __forceinline__ unsigned short f2bf(float f) {
  unsigned int u = __builtin_bit_cast(unsigned int, f);
  u += 0x7FFFu + ((u >> 16) & 1u);          // round-to-nearest-even
  return (unsigned short)(u >> 16);
}

// A-matrix bf16 16x32 fragment: lanes 0-15 row M=lane, VGPR v holds K pair
// {2v,2v+1}; lanes 16-31 add +16 to K.
__device__ __forceinline__ v16bf load_a_frag(const unsigned short* base, int stride,
                                             int row0, int k0, int lane) {
  const int r  = row0 + (lane & 15);
  const int hi = (lane >> 4) & 1;
  v8u u;
#pragma unroll
  for (int v = 0; v < 8; ++v) {
    int k = k0 + 2 * v + (hi ? 16 : 0);
    u[v] = *(const unsigned int*)(base + r * stride + k);
  }
  return __builtin_bit_cast(v16bf, u);
}

// B-matrix bf16 32x16 fragment: column N=lane&15, VGPR v holds K={2v,2v+1},
// lanes 16-31 offset K by +16. LDS tile must be stored K-pair-contiguous,
// i.e. laid out [n][k].
__device__ __forceinline__ v16bf load_b_frag(const unsigned short* base, int stride,
                                             int col0, int k0, int lane) {
  const int n  = col0 + (lane & 15);
  const int hi = (lane >> 4) & 1;
  v8u u;
#pragma unroll
  for (int v = 0; v < 8; ++v) {
    int k = k0 + 2 * v + (hi ? 16 : 0);
    u[v] = *(const unsigned int*)(base + n * stride + k);
  }
  return __builtin_bit_cast(v16bf, u);
}

__device__ __forceinline__ v8f wmma_bf16(v16bf a, v16bf b, v8f c) {
  return __builtin_amdgcn_wmma_f32_16x16x32_bf16(false, a, false, b, (short)0, c,
                                                 false, false);
}

// ---------------------------------------------------------------- cast f32->bf16
__global__ void cast_kernel(const float* __restrict__ in,
                            unsigned short* __restrict__ out, int n) {
  int i = (blockIdx.x * blockDim.x + threadIdx.x) * 4;
  int stride = gridDim.x * blockDim.x * 4;
  for (; i < n; i += stride) {
    float4 v = *(const float4*)(in + i);
    ushort4 o;
    o.x = f2bf(v.x); o.y = f2bf(v.y); o.z = f2bf(v.z); o.w = f2bf(v.w);
    *(ushort4*)(out + i) = o;
  }
}

// ---------------------------------------------------------------- layernorm -> bf16
__launch_bounds__(256)
__global__ void ln_kernel(const float* __restrict__ x, const float* __restrict__ g,
                          const float* __restrict__ bta,
                          unsigned short* __restrict__ out) {
  __shared__ float red[256];
  const int row = blockIdx.x;
  const int t = threadIdx.x;
  const float* xr = x + (size_t)row * D_MODEL;
  float4 v = *(const float4*)(xr + t * 4);

  red[t] = v.x + v.y + v.z + v.w;
  __syncthreads();
  for (int o = 128; o > 0; o >>= 1) {
    if (t < o) red[t] += red[t + o];
    __syncthreads();
  }
  const float mu = red[0] * (1.0f / D_MODEL);
  __syncthreads();

  const float dx = v.x - mu, dy = v.y - mu, dz = v.z - mu, dw = v.w - mu;
  red[t] = dx * dx + dy * dy + dz * dz + dw * dw;
  __syncthreads();
  for (int o = 128; o > 0; o >>= 1) {
    if (t < o) red[t] += red[t + o];
    __syncthreads();
  }
  const float rs = rsqrtf(red[0] * (1.0f / D_MODEL) + 1e-5f);

  const int c = t * 4;
  float4 gg = *(const float4*)(g + c);
  float4 bb = *(const float4*)(bta + c);
  unsigned short* o = out + (size_t)row * D_MODEL + c;
  o[0] = f2bf(dx * rs * gg.x + bb.x);
  o[1] = f2bf(dy * rs * gg.y + bb.y);
  o[2] = f2bf(dz * rs * gg.z + bb.z);
  o[3] = f2bf(dw * rs * gg.w + bb.w);
}

// ---------------------------------------------------------------- flash attention
// Block = 128 threads (4 waves). Each block: one (batch, head, 64-q-row) tile.
// Each wave owns a 16-row q strip -> softmax state lives in 16-lane halves.
// Mask tile is staged into LDS with coalesced int4 loads.
__launch_bounds__(128)
__global__ void attn_kernel(const unsigned short* __restrict__ Q,
                            const unsigned short* __restrict__ Km,
                            const unsigned short* __restrict__ V,
                            const int* __restrict__ mask,
                            unsigned short* __restrict__ O) {
  __shared__ unsigned short sQ[64 * 72];
  __shared__ unsigned short sK[64 * 72];     // [key][d]
  __shared__ unsigned short sVt[64 * 72];    // [d][key] (transposed)
  __shared__ unsigned short sP[4 * 16 * 72]; // per-wave P scratch [row][key]
  __shared__ int sM[64 * 64];                // mask tile [qrow][key]

  const int tid  = threadIdx.x;
  const int lane = tid & 31;
  const int wave = tid >> 5;
  const int hi   = (lane >> 4) & 1;
  const int cl   = lane & 15;
  const int q0   = blockIdx.x * 64;
  const int hoff = blockIdx.y * DK;
  const int b    = blockIdx.z;
  const size_t baseRow = (size_t)b * SEQ;
  const size_t maskBase = (size_t)b * SEQ * SEQ;
  const float scale = 0.125f; // 1/sqrt(64)

  // load 64x64 Q tile
  for (int idx = tid; idx < 64 * 16; idx += 128) {
    int r = idx >> 4;
    int c = (idx & 15) * 4;
    uint2 d = *(const uint2*)(Q + (baseRow + q0 + r) * D_MODEL + hoff + c);
    *(uint2*)(sQ + r * 72 + c) = d;
  }

  v8f acc[4];
  float mrow[8], lrow[8];
#pragma unroll
  for (int n = 0; n < 4; ++n)
#pragma unroll
    for (int j = 0; j < 8; ++j) acc[n][j] = 0.0f;
#pragma unroll
  for (int j = 0; j < 8; ++j) { mrow[j] = -3.0e38f; lrow[j] = 0.0f; }

  for (int kt = 0; kt < SEQ; kt += 64) {
    // stage K tile [key][d], V tile transposed [d][key], and mask tile
    for (int idx = tid; idx < 64 * 16; idx += 128) {
      int r = idx >> 4;
      int c = (idx & 15) * 4;
      uint2 dk = *(const uint2*)(Km + (baseRow + kt + r) * D_MODEL + hoff + c);
      *(uint2*)(sK + r * 72 + c) = dk;
      const unsigned short* vg = V + (baseRow + kt + r) * D_MODEL + hoff + c;
#pragma unroll
      for (int j = 0; j < 4; ++j) sVt[(c + j) * 72 + r] = vg[j];
      *(int4*)(sM + r * 64 + c) =
          *(const int4*)(mask + maskBase + (size_t)(q0 + r) * SEQ + kt + c);
    }
    __syncthreads();

    // S = (Q Kt) for this wave's 16-row strip, 64 key columns
    v8f s[4];
#pragma unroll
    for (int n = 0; n < 4; ++n)
#pragma unroll
      for (int j = 0; j < 8; ++j) s[n][j] = 0.0f;
#pragma unroll
    for (int kk = 0; kk < 64; kk += 32) {
      v16bf a = load_a_frag(sQ, 72, wave * 16, kk, lane);
#pragma unroll
      for (int n = 0; n < 4; ++n) {
        v16bf bf = load_b_frag(sK, 72, n * 16, kk, lane);
        s[n] = wmma_bf16(a, bf, s[n]);
      }
    }

    // scale + mask (from LDS tile)
#pragma unroll
    for (int n = 0; n < 4; ++n)
#pragma unroll
      for (int j = 0; j < 8; ++j) {
        float e = s[n][j] * scale;
        int rloc = wave * 16 + j + hi * 8;
        int ncol = n * 16 + cl;
        if (sM[rloc * 64 + ncol] == 0) e = NEGBIG;
        s[n][j] = e;
      }

    // online softmax: row max / rescale / exp / row sum (16-lane half reduction)
#pragma unroll
    for (int j = 0; j < 8; ++j) {
      float t = fmaxf(fmaxf(s[0][j], s[1][j]), fmaxf(s[2][j], s[3][j]));
      t = fmaxf(t, __shfl_xor(t, 1));
      t = fmaxf(t, __shfl_xor(t, 2));
      t = fmaxf(t, __shfl_xor(t, 4));
      t = fmaxf(t, __shfl_xor(t, 8));
      float nm = fmaxf(mrow[j], t);
      float corr = __expf(mrow[j] - nm);
      mrow[j] = nm;
      float rs = 0.0f;
#pragma unroll
      for (int n = 0; n < 4; ++n) {
        float p = __expf(s[n][j] - nm);
        s[n][j] = p;
        rs += p;
      }
      rs += __shfl_xor(rs, 1);
      rs += __shfl_xor(rs, 2);
      rs += __shfl_xor(rs, 4);
      rs += __shfl_xor(rs, 8);
      lrow[j] = lrow[j] * corr + rs;
#pragma unroll
      for (int n = 0; n < 4; ++n) acc[n][j] *= corr;
    }

    // C-layout -> A-layout transpose of P through per-wave LDS scratch
    unsigned short* pws = sP + wave * 16 * 72;
#pragma unroll
    for (int n = 0; n < 4; ++n)
#pragma unroll
      for (int j = 0; j < 8; ++j)
        pws[(j + hi * 8) * 72 + n * 16 + cl] = f2bf(s[n][j]);

    // acc += P V
#pragma unroll
    for (int kk = 0; kk < 64; kk += 32) {
      v16bf a = load_a_frag(pws, 72, 0, kk, lane);
#pragma unroll
      for (int n = 0; n < 4; ++n) {
        v16bf bf = load_b_frag(sVt, 72, n * 16, kk, lane);
        acc[n] = wmma_bf16(a, bf, acc[n]);
      }
    }
    __syncthreads();
  }

  // normalize and write O (bf16)
#pragma unroll
  for (int j = 0; j < 8; ++j) {
    float inv = 1.0f / lrow[j];
    int row = q0 + wave * 16 + j + hi * 8;
#pragma unroll
    for (int n = 0; n < 4; ++n)
      O[(baseRow + row) * D_MODEL + hoff + n * 16 + cl] = f2bf(acc[n][j] * inv);
  }
}

// ---------------------------------------------------------------- tiled GEMM
// C[M,N] = A[M,K](bf16) * W[K,N](bf16) + bias (+resid) (+relu), out f32/bf16.
// 256 threads (8 waves), tile 128x128, K-step 64 (two WMMA deep per stage).
__launch_bounds__(256)
__global__ void gemm_kernel(const unsigned short* __restrict__ A,
                            const unsigned short* __restrict__ W,
                            const float* __restrict__ bias,
                            const float* __restrict__ resid,
                            float* __restrict__ outF,
                            unsigned short* __restrict__ outB,
                            int M, int N, int K, int relu) {
  __shared__ unsigned short sA[128 * 72];  // [row][k], 64 k + pad
  __shared__ unsigned short sB[128 * 72];  // [n][k]   (transposed stage)

  const int tid  = threadIdx.x;
  const int lane = tid & 31;
  const int wave = tid >> 5;
  const int wm   = wave & 3;   // 4 waves over M (32 rows each)
  const int wn   = wave >> 2;  // 2 waves over N (64 cols each)
  const int m0   = blockIdx.x * 128;
  const int n0   = blockIdx.y * 128;

  v8f c[2][4];
#pragma unroll
  for (int mi = 0; mi < 2; ++mi)
#pragma unroll
    for (int n = 0; n < 4; ++n)
#pragma unroll
      for (int j = 0; j < 8; ++j) c[mi][n][j] = 0.0f;

  for (int k0 = 0; k0 < K; k0 += 64) {
    // A tile 128x64, row-major (uint4 = 8 bf16 per store, 16B-aligned)
#pragma unroll
    for (int i = 0; i < 4; ++i) {
      int idx = tid + i * 256;
      int r = idx >> 3;
      int cc = (idx & 7) * 8;
      uint4 d = *(const uint4*)(A + (size_t)(m0 + r) * K + k0 + cc);
      *(uint4*)(sA + r * 72 + cc) = d;
    }
    // W tile 64x128, staged transposed [n][k]
#pragma unroll
    for (int i = 0; i < 4; ++i) {
      int idx = tid + i * 256;
      int k = idx >> 4;
      int cc = (idx & 15) * 8;
      uint4 d = *(const uint4*)(W + (size_t)(k0 + k) * N + n0 + cc);
      const unsigned short* ds = (const unsigned short*)&d;
#pragma unroll
      for (int j = 0; j < 8; ++j) sB[(cc + j) * 72 + k] = ds[j];
    }
    __syncthreads();

#pragma unroll
    for (int kk = 0; kk < 64; kk += 32) {
      v16bf a0 = load_a_frag(sA, 72, wm * 32, kk, lane);
      v16bf a1 = load_a_frag(sA, 72, wm * 32 + 16, kk, lane);
#pragma unroll
      for (int n = 0; n < 4; ++n) {
        v16bf bf = load_b_frag(sB, 72, wn * 64 + n * 16, kk, lane);
        c[0][n] = wmma_bf16(a0, bf, c[0][n]);
        c[1][n] = wmma_bf16(a1, bf, c[1][n]);
      }
    }
    __syncthreads();
  }

  const int hi = (lane >> 4) & 1;
  const int cl = lane & 15;
#pragma unroll
  for (int mi = 0; mi < 2; ++mi)
#pragma unroll
    for (int n = 0; n < 4; ++n)
#pragma unroll
      for (int j = 0; j < 8; ++j) {
        int row = m0 + wm * 32 + mi * 16 + j + hi * 8;
        int col = n0 + wn * 64 + n * 16 + cl;
        float v = c[mi][n][j] + bias[col];
        if (relu) v = fmaxf(v, 0.0f);
        size_t off = (size_t)row * N + col;
        if (resid) v += resid[off];
        if (outF) outF[off] = v;
        if (outB) outB[off] = f2bf(v);
      }
}

// ---------------------------------------------------------------- launch
extern "C" void kernel_launch(void* const* d_in, const int* in_sizes, int n_in,
                              void* d_out, int out_size, void* d_ws, size_t ws_size,
                              hipStream_t stream) {
  const float* tgt      = (const float*)d_in[0];
  const float* memory   = (const float*)d_in[1];
  const int*   tgt_mask = (const int*)d_in[2];
  const int*   mem_mask = (const int*)d_in[3];
  const float* sa_w  = (const float*)d_in[4];
  const float* sa_b  = (const float*)d_in[5];
  const float* mha_w = (const float*)d_in[6];
  const float* mha_b = (const float*)d_in[7];
  const float* ff_w1 = (const float*)d_in[8];
  const float* ff_b1 = (const float*)d_in[9];
  const float* ff_w2 = (const float*)d_in[10];
  const float* ff_b2 = (const float*)d_in[11];
  const float* ln1_g = (const float*)d_in[12];
  const float* ln1_b = (const float*)d_in[13];
  const float* ln2_g = (const float*)d_in[14];
  const float* ln2_b = (const float*)d_in[15];
  const float* ln3_g = (const float*)d_in[16];
  const float* ln3_b = (const float*)d_in[17];
  float* out = (float*)d_out;

  char* ws = (char*)d_ws;
  const size_t MiB = (size_t)1 << 20;
  unsigned short* wsa  = (unsigned short*)(ws + 0 * MiB);   // 2 MiB
  unsigned short* wmha = (unsigned short*)(ws + 2 * MiB);   // 2 MiB
  unsigned short* wff1 = (unsigned short*)(ws + 4 * MiB);   // 8 MiB
  unsigned short* wff2 = (unsigned short*)(ws + 12 * MiB);  // 8 MiB
  unsigned short* memb = (unsigned short*)(ws + 20 * MiB);  // 8 MiB
  unsigned short* hb   = (unsigned short*)(ws + 28 * MiB);  // 8 MiB
  unsigned short* ob   = (unsigned short*)(ws + 36 * MiB);  // 8 MiB
  float*          x1   = (float*)(ws + 44 * MiB);           // 16 MiB
  unsigned short* tb   = (unsigned short*)(ws + 60 * MiB);  // 32 MiB

  const int ROWS = BATCH * SEQ;  // 4096
  const dim3 B256(256), B128(128);
  const dim3 attnGrid(SEQ / 64, N_HEAD, BATCH);
  const dim3 gemmD(ROWS / 128, D_MODEL / 128);   // 32 x 8
  const dim3 gemmF(ROWS / 128, D_FF / 128);      // 32 x 32

  // one-time casts to bf16
  cast_kernel<<<1024, B256, 0, stream>>>(sa_w,  wsa,  D_MODEL * D_MODEL);
  cast_kernel<<<1024, B256, 0, stream>>>(mha_w, wmha, D_MODEL * D_MODEL);
  cast_kernel<<<1024, B256, 0, stream>>>(ff_w1, wff1, D_MODEL * D_FF);
  cast_kernel<<<1024, B256, 0, stream>>>(ff_w2, wff2, D_FF * D_MODEL);
  cast_kernel<<<1024, B256, 0, stream>>>(memory, memb, ROWS * D_MODEL);

  // ---- self-attention block: x1 = tgt + Attn(LN1(tgt)) @ sa_w + sa_b
  ln_kernel<<<ROWS, B256, 0, stream>>>(tgt, ln1_g, ln1_b, hb);
  attn_kernel<<<attnGrid, B128, 0, stream>>>(hb, hb, hb, tgt_mask, ob);
  gemm_kernel<<<gemmD, B256, 0, stream>>>(ob, wsa, sa_b, tgt, x1, nullptr,
                                          ROWS, D_MODEL, D_MODEL, 0);

  // ---- cross-attention: q=k=memory, v=LN2(x1); x2 = x1 + O @ mha_w + mha_b
  ln_kernel<<<ROWS, B256, 0, stream>>>(x1, ln2_g, ln2_b, hb);
  attn_kernel<<<attnGrid, B128, 0, stream>>>(memb, memb, hb, mem_mask, ob);
  gemm_kernel<<<gemmD, B256, 0, stream>>>(ob, wmha, mha_b, x1, out, nullptr,
                                          ROWS, D_MODEL, D_MODEL, 0);

  // ---- feed-forward: out = x2 + relu(LN3(x2) @ w1 + b1) @ w2 + b2
  ln_kernel<<<ROWS, B256, 0, stream>>>(out, ln3_g, ln3_b, hb);
  gemm_kernel<<<gemmF, B256, 0, stream>>>(hb, wff1, ff_b1, nullptr, nullptr, tb,
                                          ROWS, D_FF, D_MODEL, 1);
  gemm_kernel<<<gemmD, B256, 0, stream>>>(tb, wff2, ff_b2, out, out, nullptr,
                                          ROWS, D_MODEL, D_FF, 0);
}
